// MultiHeadAttention_2869038154325
// MI455X (gfx1250) — matrix operand order repaired and checked by reading, not compiled, not run on hardware
//
#include <hip/hip_runtime.h>
#include <hip/hip_bf16.h>
#include <math.h>

#define D_MODEL   1024
#define NUM_HEADS 16
#define HEAD_DIM  64
#define BATCH     2
#define SEQ       2048
#define QKV_N     (3 * D_MODEL)

typedef __attribute__((ext_vector_type(16))) __bf16 v16bf;
typedef __attribute__((ext_vector_type(8)))  float  v8f;
typedef __attribute__((ext_vector_type(8)))  __bf16 bf16x8;
typedef __attribute__((ext_vector_type(4)))  int    v4i;

// ---------------------------------------------------------------------------
// CDNA5 async global->LDS staging (builtin confirmed:
//   (v4i addrspace(1)*, v4i addrspace(3)*, imm offset, imm cpol))
// ---------------------------------------------------------------------------
#define GLOBAL_AS __attribute__((address_space(1)))
#define LDS_AS    __attribute__((address_space(3)))

#if defined(__AMDGCN__) && __has_builtin(__builtin_amdgcn_global_load_async_to_lds_b128)
#define HAVE_ASYNC_LDS 1
#define ASYNC_COPY_B128(gp, lp)                                              \
  __builtin_amdgcn_global_load_async_to_lds_b128(                            \
      (GLOBAL_AS v4i*)(void*)(gp), (LDS_AS v4i*)(void*)(lp), 0, 0)
#endif

static __device__ __forceinline__ void wait_async_all() {
#if defined(__AMDGCN__)
#if __has_builtin(__builtin_amdgcn_s_wait_asynccnt)
  __builtin_amdgcn_s_wait_asynccnt(0);
#else
  asm volatile("s_wait_asynccnt 0x0" ::: "memory");
#endif
#endif
}

static __device__ __forceinline__ float fast_rcp(float x) {
#if defined(__AMDGCN__) && __has_builtin(__builtin_amdgcn_rcpf)
  return __builtin_amdgcn_rcpf(x);
#else
  return 1.0f / x;
#endif
}

// ---------------------------------------------------------------------------
// fp32 -> bf16 conversion (one-time)
// ---------------------------------------------------------------------------
__global__ void cvt_f32_to_bf16(const float* __restrict__ in,
                                __bf16* __restrict__ out, int n) {
  int i = blockIdx.x * blockDim.x + threadIdx.x;
  int stride = gridDim.x * blockDim.x;
  for (; i < n; i += stride) out[i] = (__bf16)in[i];
}

// ---------------------------------------------------------------------------
// Tiled bf16 GEMM:  C[M,N] = A[M,K] * B[K,N] + bias[N]
// Block tile 128x128, BK=32; 8 waves (2Mx4N), each wave 64x32 = 4x2 WMMA
// tiles. Double-buffered LDS pipeline with async A copies.
// ---------------------------------------------------------------------------
template <typename OutT>
__global__ __launch_bounds__(256)
void gemm_bf16(const __bf16* __restrict__ A, const __bf16* __restrict__ B,
               const float* __restrict__ bias, OutT* __restrict__ C,
               int M, int N, int K) {
  __shared__ __bf16 As[2][128][40];   // [buf][m][k]
  __shared__ __bf16 Bst[2][128][40];  // [buf][n][k] transposed

  const int tid  = threadIdx.x;
  const int lane = tid & 31;
  const int w    = tid >> 5;
  const int wm   = w >> 2;  // 0..1 (M)
  const int wn   = w & 3;   // 0..3 (N)
  const int l15  = lane & 15;
  const int lh   = lane >> 4;

  const int rowBase = blockIdx.y * 128;
  const int colBase = blockIdx.x * 128;

  bf16x8 vbreg[2];

  auto stage_A_async = [&](int buf, int k0) {
#pragma unroll
    for (int it = 0; it < 2; ++it) {
      int chunk = tid + it * 256;       // 0..511
      int row = chunk >> 2;             // 0..127
      int kc  = (chunk & 3) * 8;        // 0,8,16,24
      const __bf16* ga = A + (size_t)(rowBase + row) * K + k0 + kc;
#ifdef HAVE_ASYNC_LDS
      ASYNC_COPY_B128(ga, &As[buf][row][kc]);
#else
      *(bf16x8*)(&As[buf][row][kc]) = *(const bf16x8*)ga;
#endif
    }
  };
  auto load_B = [&](int k0) {
#pragma unroll
    for (int it = 0; it < 2; ++it) {
      int chunk = tid + it * 256;       // 0..511
      int kr = chunk >> 4;              // 0..31
      int nc = (chunk & 15) * 8;        // 0..120
      vbreg[it] = *(const bf16x8*)(B + (size_t)(k0 + kr) * N + colBase + nc);
    }
  };
  auto store_B = [&](int buf) {
#pragma unroll
    for (int it = 0; it < 2; ++it) {
      int chunk = tid + it * 256;
      int kr = chunk >> 4;
      int nc = (chunk & 15) * 8;
#pragma unroll
      for (int j = 0; j < 8; ++j) Bst[buf][nc + j][kr] = vbreg[it][j];
    }
  };

  v8f acc[4][2];
#pragma unroll
  for (int mt = 0; mt < 4; ++mt)
#pragma unroll
    for (int nt = 0; nt < 2; ++nt) acc[mt][nt] = (v8f)0.0f;

  // prologue: stage tile 0
  stage_A_async(0, 0);
  load_B(0);
  store_B(0);
  wait_async_all();
  __syncthreads();

  int buf = 0;
  for (int k0 = 0; k0 < K; k0 += 32, buf ^= 1) {
    const bool has_next = (k0 + 32 < K);
    if (has_next) {
      stage_A_async(buf ^ 1, k0 + 32);  // async: lands during compute
      load_B(k0 + 32);                  // global loads in flight over WMMAs
    }

    // ---- pipelined fragment loads + 8 WMMAs ----
    // B fragments (2) loaded up front; A fragments rotate through 2 slots so
    // the loads for step i+1 are in flight while WMMA i executes.
    v16bf bfr[2];
#pragma unroll
    for (int nt = 0; nt < 2; ++nt) {
      int n = wn * 32 + nt * 16 + l15;
#pragma unroll
      for (int v = 0; v < 8; ++v) {
        int kk = (v < 4 ? 2 * v : 16 + 2 * (v - 4)) + lh * 8;
        bfr[nt][2 * v]     = Bst[buf][n][kk];
        bfr[nt][2 * v + 1] = Bst[buf][n][kk + 1];
      }
    }
    v16bf af[2];
    auto load_af = [&](int slot, int mt) {
      int m = wm * 64 + mt * 16 + l15;
#pragma unroll
      for (int v = 0; v < 8; ++v) {
        int kp = (v < 4 ? 2 * v : 16 + 2 * (v - 4)) + lh * 8;
        af[slot][2 * v]     = As[buf][m][kp];
        af[slot][2 * v + 1] = As[buf][m][kp + 1];
      }
    };
    load_af(0, 0);
#pragma unroll
    for (int mt = 0; mt < 4; ++mt) {
      if (mt < 3) load_af((mt + 1) & 1, mt + 1);
#pragma unroll
      for (int nt = 0; nt < 2; ++nt)
        acc[mt][nt] = __builtin_amdgcn_wmma_f32_16x16x32_bf16(
            false, af[mt & 1], false, bfr[nt], (short)0, acc[mt][nt], false,
            false);
    }

    if (has_next) store_B(buf ^ 1);
    wait_async_all();
    __syncthreads();
  }

  // ---- epilogue: bias add, store ----
#pragma unroll
  for (int mt = 0; mt < 4; ++mt) {
#pragma unroll
    for (int nt = 0; nt < 2; ++nt) {
      int col = colBase + wn * 32 + nt * 16 + l15;
      float bv = bias[col];
#pragma unroll
      for (int r = 0; r < 8; ++r) {
        int row = rowBase + wm * 64 + mt * 16 + r + 8 * lh;
        C[(size_t)row * N + col] = (OutT)(acc[mt][nt][r] + bv);
      }
    }
  }
}

// ---------------------------------------------------------------------------
// Flash attention: grid (S/128, H, B), 256 threads = 8 waves, 64-key slabs.
// Double-buffered K/V staging; score and PV loops explicitly software-
// pipelined with two-slot B-fragment rotation so LDS latency hides under
// the in-flight WMMA instead of a dscnt-0 stall per matrix op.
// ---------------------------------------------------------------------------
__global__ __launch_bounds__(256)
void attn_kernel(const __bf16* __restrict__ qkv, const float* __restrict__ mask,
                 __bf16* __restrict__ vals) {
  const int S  = SEQ;
  const int hd = HEAD_DIM;
  const int ld = QKV_N;  // 3072

  const int b = blockIdx.z;
  const int h = blockIdx.y;
  const int tid  = threadIdx.x;
  const int lane = tid & 31;
  const int w    = tid >> 5;
  const int l15  = lane & 15;
  const int lh   = lane >> 4;
  const int qrow0 = blockIdx.x * 128 + w * 16;

  __shared__ __bf16 Ks[2][64][72];   // [buf][key][d]  (B-operand for Q*K^T)
  __shared__ __bf16 Vs[2][64][72];   // [buf][d][key]  (B-operand for P*V)
  __shared__ __bf16 Ps[8][16][72];   // per-wave P bounce buffer

  const __bf16* qbase = qkv + (size_t)b * S * ld + h * (3 * hd);

  bf16x8 vvreg[2];

  auto stage_K_async = [&](int buf, int kt) {
#pragma unroll
    for (int it = 0; it < 2; ++it) {
      int chunk = tid + it * 256;    // 0..511
      int key = chunk >> 3;          // 0..63
      int dc  = (chunk & 7) * 8;     // 0..56
      const __bf16* gk = qbase + (size_t)(kt + key) * ld + hd + dc;
#ifdef HAVE_ASYNC_LDS
      ASYNC_COPY_B128(gk, &Ks[buf][key][dc]);
#else
      *(bf16x8*)(&Ks[buf][key][dc]) = *(const bf16x8*)gk;
#endif
    }
  };
  auto load_V = [&](int kt) {
#pragma unroll
    for (int it = 0; it < 2; ++it) {
      int chunk = tid + it * 256;
      int key = chunk >> 3;
      int dc  = (chunk & 7) * 8;
      vvreg[it] = *(const bf16x8*)(qbase + (size_t)(kt + key) * ld + 2 * hd + dc);
    }
  };
  auto store_V = [&](int buf) {
#pragma unroll
    for (int it = 0; it < 2; ++it) {
      int chunk = tid + it * 256;
      int key = chunk >> 3;
      int dc  = (chunk & 7) * 8;
#pragma unroll
      for (int j = 0; j < 8; ++j) Vs[buf][dc + j][key] = vvreg[it][j];
    }
  };

  // Q fragments (A layout, d split 0..31/32..63), 1/sqrt(hd) folded in
  const float scale = 0.125f;
  v16bf qf[2];
#pragma unroll
  for (int df = 0; df < 2; ++df) {
    int m = qrow0 + l15;
#pragma unroll
    for (int v = 0; v < 8; ++v) {
      int kp = (v < 4 ? 2 * v : 16 + 2 * (v - 4)) + lh * 8;
      int d  = df * 32 + kp;
      qf[df][2 * v]     = (__bf16)(scale * (float)qbase[(size_t)m * ld + d]);
      qf[df][2 * v + 1] = (__bf16)(scale * (float)qbase[(size_t)m * ld + d + 1]);
    }
  }

  v8f acco[4];
#pragma unroll
  for (int dt = 0; dt < 4; ++dt) acco[dt] = (v8f)0.0f;
  float m_run[8], l_run[8];
#pragma unroll
  for (int r = 0; r < 8; ++r) { m_run[r] = -1e30f; l_run[r] = 0.0f; }

  const float* mrow = mask + (size_t)b * S * S;

  // prologue: stage slab 0
  stage_K_async(0, 0);
  load_V(0);
  store_V(0);
  wait_async_all();
  __syncthreads();

  int buf = 0;
  for (int kt = 0; kt < S; kt += 64, buf ^= 1) {
    const bool has_next = (kt + 64 < S);
    if (has_next) {
      stage_K_async(buf ^ 1, kt + 64);
      load_V(kt + 64);
    }

    // ---- scores: 8 WMMAs (nt=key tile, df=d half), pipelined B frags ----
    v16bf bb[2];
    auto load_bb = [&](int slot, int nt, int df) {
      int n = nt * 16 + l15;
#pragma unroll
      for (int v = 0; v < 8; ++v) {
        int kk = df * 32 + (v < 4 ? 2 * v : 16 + 2 * (v - 4)) + lh * 8;
        bb[slot][2 * v]     = Ks[buf][n][kk];
        bb[slot][2 * v + 1] = Ks[buf][n][kk + 1];
      }
    };
    v8f sc[4];
    load_bb(0, 0, 0);
#pragma unroll
    for (int i = 0; i < 8; ++i) {
      int nt = i >> 1, df = i & 1;
      if (i < 7) load_bb((i + 1) & 1, (i + 1) >> 1, (i + 1) & 1);
      v8f cin = (df == 0) ? (v8f)0.0f : sc[nt];
      sc[nt] = __builtin_amdgcn_wmma_f32_16x16x32_bf16(
          false, qf[df], false, bb[i & 1], (short)0, cin, false, false);
    }

    // ---- online softmax over the 64-key slab ----
#pragma unroll
    for (int r = 0; r < 8; ++r) {
      int row = qrow0 + r + 8 * lh;
      const float* mp = mrow + (size_t)row * S + kt + l15;
      float s0 = sc[0][r] + mp[0];
      float s1 = sc[1][r] + mp[16];
      float s2 = sc[2][r] + mp[32];
      float s3 = sc[3][r] + mp[48];
      float mx = fmaxf(fmaxf(s0, s1), fmaxf(s2, s3));
#pragma unroll
      for (int off = 8; off >= 1; off >>= 1)
        mx = fmaxf(mx, __shfl_xor(mx, off, 32));
      float mnew  = fmaxf(m_run[r], mx);
      float alpha = __expf(m_run[r] - mnew);
      float p0 = __expf(s0 - mnew);
      float p1 = __expf(s1 - mnew);
      float p2 = __expf(s2 - mnew);
      float p3 = __expf(s3 - mnew);
      float rs = (p0 + p1) + (p2 + p3);
#pragma unroll
      for (int off = 8; off >= 1; off >>= 1) rs += __shfl_xor(rs, off, 32);
      l_run[r] = l_run[r] * alpha + rs;
      m_run[r] = mnew;
#pragma unroll
      for (int dt = 0; dt < 4; ++dt) acco[dt][r] *= alpha;
      int pr = r + 8 * lh;
      Ps[w][pr][l15]      = (__bf16)p0;
      Ps[w][pr][16 + l15] = (__bf16)p1;
      Ps[w][pr][32 + l15] = (__bf16)p2;
      Ps[w][pr][48 + l15] = (__bf16)p3;
    }

    // ---- P x V: 8 WMMAs (dt=d tile, kh2=key half), pipelined B frags ----
    v16bf pf[2];
#pragma unroll
    for (int kh2 = 0; kh2 < 2; ++kh2) {
#pragma unroll
      for (int v = 0; v < 8; ++v) {
        int kp = (v < 4 ? 2 * v : 16 + 2 * (v - 4)) + lh * 8;
        pf[kh2][2 * v]     = Ps[w][l15][kh2 * 32 + kp];
        pf[kh2][2 * v + 1] = Ps[w][l15][kh2 * 32 + kp + 1];
      }
    }
    v16bf bv[2];
    auto load_bv = [&](int slot, int dt, int kh2) {
      int n = dt * 16 + l15;
#pragma unroll
      for (int v = 0; v < 8; ++v) {
        int kk = kh2 * 32 + (v < 4 ? 2 * v : 16 + 2 * (v - 4)) + lh * 8;
        bv[slot][2 * v]     = Vs[buf][n][kk];
        bv[slot][2 * v + 1] = Vs[buf][n][kk + 1];
      }
    };
    load_bv(0, 0, 0);
#pragma unroll
    for (int i = 0; i < 8; ++i) {
      int dt = i >> 1, kh2 = i & 1;
      if (i < 7) load_bv((i + 1) & 1, (i + 1) >> 1, (i + 1) & 1);
      acco[dt] = __builtin_amdgcn_wmma_f32_16x16x32_bf16(
          false, pf[kh2], false, bv[i & 1], (short)0, acco[dt], false, false);
    }

    if (has_next) store_V(buf ^ 1);
    wait_async_all();
    __syncthreads();
  }

  // ---- normalize and write vals[B*S, D] (bf16) ----
#pragma unroll
  for (int dt = 0; dt < 4; ++dt) {
#pragma unroll
    for (int r = 0; r < 8; ++r) {
      int row = qrow0 + r + 8 * lh;
      float o = acco[dt][r] * fast_rcp(l_run[r]);
      vals[(size_t)(b * S + row) * D_MODEL + h * hd + dt * 16 + l15] = (__bf16)o;
    }
  }
}

// ---------------------------------------------------------------------------
// Launch
// ---------------------------------------------------------------------------
extern "C" void kernel_launch(void* const* d_in, const int* in_sizes, int n_in,
                              void* d_out, int out_size, void* d_ws,
                              size_t ws_size, hipStream_t stream) {
  const float* x    = (const float*)d_in[0];
  const float* mask = (const float*)d_in[1];
  const float* Wqkv = (const float*)d_in[2];
  const float* bqkv = (const float*)d_in[3];
  const float* Wo   = (const float*)d_in[4];
  const float* bo   = (const float*)d_in[5];
  float* out = (float*)d_out;

  const int M = BATCH * SEQ;  // 4096

  char* ws = (char*)d_ws;
  size_t off = 0;
  __bf16* xb    = (__bf16*)(ws + off); off += (size_t)M * D_MODEL * 2;
  __bf16* wqkvb = (__bf16*)(ws + off); off += (size_t)D_MODEL * QKV_N * 2;
  __bf16* wob   = (__bf16*)(ws + off); off += (size_t)D_MODEL * D_MODEL * 2;
  __bf16* qkvb  = (__bf16*)(ws + off); off += (size_t)M * QKV_N * 2;
  __bf16* valsb = (__bf16*)(ws + off); off += (size_t)M * D_MODEL * 2;
  (void)ws_size; (void)in_sizes; (void)n_in; (void)out_size;

  cvt_f32_to_bf16<<<1024, 256, 0, stream>>>(x, xb, M * D_MODEL);
  cvt_f32_to_bf16<<<1024, 256, 0, stream>>>(Wqkv, wqkvb, D_MODEL * QKV_N);
  cvt_f32_to_bf16<<<512, 256, 0, stream>>>(Wo, wob, D_MODEL * D_MODEL);

  // QKV projection: [4096,1024] x [1024,3072] + bqkv -> bf16
  gemm_bf16<__bf16><<<dim3(QKV_N / 128, M / 128), 256, 0, stream>>>(
      xb, wqkvb, bqkv, qkvb, M, QKV_N, D_MODEL);

  // Attention
  attn_kernel<<<dim3(SEQ / 128, NUM_HEADS, BATCH), 256, 0, stream>>>(
      qkvb, mask, valsb);

  // Output projection: [4096,1024] x [1024,1024] + bo -> fp32 out
  gemm_bf16<float><<<dim3(D_MODEL / 128, M / 128), 256, 0, stream>>>(
      valsb, wob, bo, out, M, D_MODEL, D_MODEL);
}
